// SelfAttention_16973710754186
// MI455X (gfx1250) — matrix operand order repaired and checked by reading, not compiled, not run on hardware
//
#include <hip/hip_runtime.h>

// out = x @ kernel[2]  (softmax over a length-1 axis == 1, so attention == V projection)
// M=8192, N=1024, K=1024, fp32 in/out, f16 WMMA with f32 accumulate.
// Double-buffered LDS pipeline: 1 barrier per K-step, global loads overlap WMMA.

typedef __attribute__((ext_vector_type(2)))  _Float16 v2h;
typedef __attribute__((ext_vector_type(4)))  _Float16 v4h;
typedef __attribute__((ext_vector_type(8)))  _Float16 v8h;
typedef __attribute__((ext_vector_type(16))) _Float16 v16h;
typedef __attribute__((ext_vector_type(8)))  float    v8f;

#define DIN  1024
#define DOUT 1024
#define BM   128
#define BN   128
#define BK   32
#define NT   (DIN / BK)   // 32 K-steps
#define LDA  48   // padded stride (halves) per A row: multiple of 8 (16B-aligned b128), != 32 (bank spread)
#define LDB  48   // padded stride (halves) per B column

// Build a 16-half fragment from two 16-byte LDS chunks.
__device__ __forceinline__ v16h load_frag(const _Float16* p0, const _Float16* p1) {
    v8h lo = *(const v8h*)p0;
    v8h hi = *(const v8h*)p1;
    v16h r;
#pragma unroll
    for (int i = 0; i < 8; ++i) { r[i] = lo[i]; r[i + 8] = hi[i]; }
    return r;
}

// Stage one 128x32 A tile + 32x128 B tile (fp32) from global into registers.
__device__ __forceinline__ void load_tiles(const float* __restrict__ x,
                                           const float* __restrict__ w,
                                           int bm, int bn, int kt, int tid,
                                           float4 (&ar)[4], float4 (&br)[4]) {
#pragma unroll
    for (int i = 0; i < 4; ++i) {
        const int ai   = tid + i * 256;      // 1024 float4 slots of the 128x32 A tile
        const int arow = ai >> 3;            // 0..127
        const int ac4  = ai & 7;             // float4 index within 32-wide K chunk
        ar[i] = *(const float4*)(x + (size_t)(bm + arow) * DIN + kt + ac4 * 4);
    }
#pragma unroll
    for (int i = 0; i < 2; ++i) {
        // 512 (k-pair, n4) items: each loads two float4 from K-adjacent rows, same N cols
        const int bi = tid + i * 256;
        const int kp = bi >> 5;              // 0..15  -> K rows 2kp, 2kp+1
        const int n4 = bi & 31;              // 0..31  -> N cols n4*4 .. n4*4+3
        br[2 * i + 0] = *(const float4*)(w + (size_t)(kt + 2 * kp + 0) * DOUT + bn + n4 * 4);
        br[2 * i + 1] = *(const float4*)(w + (size_t)(kt + 2 * kp + 1) * DOUT + bn + n4 * 4);
    }
}

// Convert f32 -> f16 and store the staged tiles into an LDS buffer.
__device__ __forceinline__ void store_tiles(_Float16* __restrict__ As,
                                            _Float16* __restrict__ Bs,
                                            int tid,
                                            const float4 (&ar)[4], const float4 (&br)[4]) {
#pragma unroll
    for (int i = 0; i < 4; ++i) {
        const int ai   = tid + i * 256;
        const int arow = ai >> 3;
        const int ac4  = ai & 7;
        v4h ah;
        ah[0] = (_Float16)ar[i].x; ah[1] = (_Float16)ar[i].y;
        ah[2] = (_Float16)ar[i].z; ah[3] = (_Float16)ar[i].w;
        *(v4h*)(&As[arow * LDA + ac4 * 4]) = ah;           // ds_store_b64
    }
#pragma unroll
    for (int i = 0; i < 2; ++i) {
        const int bi = tid + i * 256;
        const int kp = bi >> 5;
        const int n4 = bi & 31;
        const float4 b0 = br[2 * i + 0];                   // K row 2kp
        const float4 b1 = br[2 * i + 1];                   // K row 2kp+1
        const float f0[4] = {b0.x, b0.y, b0.z, b0.w};
        const float f1[4] = {b1.x, b1.y, b1.z, b1.w};
#pragma unroll
        for (int j = 0; j < 4; ++j) {
            v2h p; p[0] = (_Float16)f0[j]; p[1] = (_Float16)f1[j];
            // column-major Bs[n][k]: K-adjacent pair -> one ds_store_b32
            *(v2h*)(&Bs[(n4 * 4 + j) * LDB + 2 * kp]) = p;
        }
    }
}

__global__ __launch_bounds__(256)
void vproj_gemm_wmma(const float* __restrict__ x,
                     const float* __restrict__ w,     // kernel[2], row-major [K=1024][N=1024]
                     float* __restrict__ out) {
    __shared__ __align__(16) _Float16 As[2][BM * LDA];   // row-major rows of x tile (f16)
    __shared__ __align__(16) _Float16 Bs[2][BN * LDB];   // column-major cols of W tile (f16)

    const int tid   = threadIdx.x;
    const int lane  = tid & 31;
    const int wave  = tid >> 5;
    const int wm    = (wave >> 2) * 64;   // wave row offset inside C tile (2 waves in M)
    const int wn    = (wave & 3)  * 32;   // wave col offset inside C tile (4 waves in N)
    const int bm    = blockIdx.y * BM;
    const int bn    = blockIdx.x * BN;
    const int lhalf = lane & 15;
    const bool hi16 = lane >= 16;

    v8f acc[4][2];
#pragma unroll
    for (int mt = 0; mt < 4; ++mt)
#pragma unroll
        for (int nt = 0; nt < 2; ++nt)
            acc[mt][nt] = (v8f){0.f, 0.f, 0.f, 0.f, 0.f, 0.f, 0.f, 0.f};

    float4 ar[4], br[4];

    // Prologue: stage tile 0 into LDS buffer 0.
    load_tiles(x, w, bm, bn, 0, tid, ar, br);
    store_tiles(As[0], Bs[0], tid, ar, br);
    __syncthreads();

    for (int t = 0; t < NT; ++t) {
        // Kick off global loads for tile t+1 while tile t computes.
        if (t + 1 < NT)
            load_tiles(x, w, bm, bn, (t + 1) * BK, tid, ar, br);

        const _Float16* Ab = As[t & 1];
        const _Float16* Bb = Bs[t & 1];

        // ---- B fragments: lane<16 holds col N=lane, K=0..15; lane>=16 holds K=16..31 ----
        v16h bfrag[2];
#pragma unroll
        for (int nt = 0; nt < 2; ++nt) {
            const int col = wn + nt * 16 + lhalf;
            const int kb  = hi16 ? 16 : 0;
            bfrag[nt] = load_frag(&Bb[col * LDB + kb], &Bb[col * LDB + kb + 8]);
        }

        // ---- A fragments + WMMA ----
#pragma unroll
        for (int mt = 0; mt < 4; ++mt) {
            const int row = wm + mt * 16 + lhalf;
            const int ka  = hi16 ? 16 : 0;  // lanes 0-15: K 0..15; lanes 16-31: K 16..31
            v16h afrag = load_frag(&Ab[row * LDA + ka], &Ab[row * LDA + ka + 8]);
#pragma unroll
            for (int nt = 0; nt < 2; ++nt) {
                acc[mt][nt] = __builtin_amdgcn_wmma_f32_16x16x32_f16(
                    /*neg_a=*/false, afrag,
                    /*neg_b=*/false, bfrag[nt],
                    /*c_mod=*/(short)0, acc[mt][nt],
                    /*reuse_a=*/false, /*reuse_b=*/false);
            }
        }

        // Store tile t+1 into the other buffer; single barrier per iteration.
        if (t + 1 < NT) {
            store_tiles(As[(t + 1) & 1], Bs[(t + 1) & 1], tid, ar, br);
            __syncthreads();
        }
    }

    // ---- epilogue: C/D layout -> global f32 ----
#pragma unroll
    for (int mt = 0; mt < 4; ++mt) {
#pragma unroll
        for (int nt = 0; nt < 2; ++nt) {
            const int colBase = bn + wn + nt * 16 + lhalf;
#pragma unroll
            for (int r = 0; r < 8; ++r) {
                const int row = bm + wm + mt * 16 + (hi16 ? r + 8 : r);
                out[(size_t)row * DOUT + colBase] = acc[mt][nt][r];
            }
        }
    }
}

extern "C" void kernel_launch(void* const* d_in, const int* in_sizes, int n_in,
                              void* d_out, int out_size, void* d_ws, size_t ws_size,
                              hipStream_t stream) {
    const float* x      = (const float*)d_in[0];                    // (8192, 1024)
    const float* kernel = (const float*)d_in[1];                    // (3, 1024, 1024)
    const float* wv     = kernel + (size_t)2 * DIN * DOUT;          // kernel[2]
    float* out          = (float*)d_out;                            // (8192, 1, 1024)

    dim3 grid(DOUT / BN, 8192 / BM);   // (8, 64)
    dim3 block(256);
    vproj_gemm_wmma<<<grid, block, 0, stream>>>(x, wv, out);
}
